// LocalWindowAttention_61675730370607
// MI455X (gfx1250) — matrix operand order repaired and checked by reading, not compiled, not run on hardware
//
#include <hip/hip_runtime.h>
#include <hip/hip_bf16.h>

// ---------------------------------------------------------------------------
// Local window attention (GQA), CDNA5 / gfx1250, bf16 WMMA pipeline.
//   B=2, T=4096, D=2048, H=16, HK=4, HD=128, WIN=512
// Stages:
//   1) Q = x@Wq, K = x@Wk, V = x@Wv   (f32 in -> bf16 out, WMMA bf16)
//   2) flash-style windowed attention (WMMA bf16, f32 softmax)  -> O bf16
//      K tile staged by the Tensor Data Mover when available.
//   3) out = O@Wo                      (bf16 in -> f32 out, WMMA bf16)
//      A tile staged by async global->LDS loads when available.
// ---------------------------------------------------------------------------

typedef __attribute__((ext_vector_type(16))) __bf16 v16bf;
typedef __attribute__((ext_vector_type(8)))  float  v8f;
typedef __attribute__((ext_vector_type(4)))  unsigned u32x4;
typedef __attribute__((ext_vector_type(8)))  int      i32x8;
typedef __attribute__((ext_vector_type(4)))  int      i32x4;

// gcc-style vector for the async-LDS builtin's pointer params
typedef int v4i_g __attribute__((vector_size(16)));
typedef __attribute__((address_space(1))) v4i_g GlobV4;
typedef __attribute__((address_space(3))) v4i_g LdsV4;

union BFrag { v16bf v; uint4 u[2]; };

constexpr int Hh  = 16;    // query heads
constexpr int HKk = 4;     // kv heads
constexpr int HD  = 128;   // head dim
constexpr int WIN = 512;   // window
constexpr int Bb  = 2;
constexpr int Tt  = 4096;
constexpr int Dd  = 2048;
constexpr int MTOT = Bb * Tt;   // 8192

#define HAS_TDM        (__has_builtin(__builtin_amdgcn_tensor_load_to_lds))
#define HAS_ASYNC_B128 (__has_builtin(__builtin_amdgcn_global_load_async_to_lds_b128))
#define HAS_WAIT_ASYNC (__has_builtin(__builtin_amdgcn_s_wait_asynccnt))

__device__ inline v16bf load_frag(const __bf16* p0, const __bf16* p1) {
  BFrag f;
  f.u[0] = *(const uint4*)p0;   // ds_load_b128 / global_load_b128
  f.u[1] = *(const uint4*)p1;
  return f.v;
}

__device__ inline v8f wmma_bf16(v16bf a, v16bf b, v8f c) {
  // D = A(16x32) * B(32x16) + C, f32 accumulate
  return __builtin_amdgcn_wmma_f32_16x16x32_bf16(
      /*neg_a=*/false, a, /*neg_b=*/false, b,
      /*c_mod=*/(short)0, c, /*reuse_a=*/false, /*reuse_b=*/false);
}

#if HAS_TDM
// TDM 2D tile load: bf16 elements (data_size code 1 = 2 bytes).
// Descriptor packing per CDNA5 ISA ch.8 (D# group 0 / group 1).
__device__ inline void tdm_load_2d_bf16(unsigned lds_addr, const void* gaddr,
                                        unsigned tile_d0, unsigned tile_d1,
                                        unsigned tensor_d0, unsigned tensor_d1,
                                        unsigned stride_d0) {
  const unsigned long long ga = (unsigned long long)(size_t)gaddr;
  u32x4 g0;
  g0[0] = 1u;                                            // count=1, user mode
  g0[1] = lds_addr;                                      // LDS byte address
  g0[2] = (unsigned)(ga & 0xffffffffu);                  // global addr [31:0]
  g0[3] = (unsigned)((ga >> 32) & 0x1ffffffu) | (2u << 30);  // [56:32]|type=2
  i32x8 g1;
  g1[0] = (int)(1u << 16);                               // data_size=1 (2B)
  g1[1] = (int)(tensor_d0 << 16);                        // tensor_dim0[15:0]
  g1[2] = (int)((tensor_d0 >> 16) | (tensor_d1 << 16));  // dim0 hi | dim1 lo
  g1[3] = (int)((tensor_d1 >> 16) | (tile_d0 << 16));    // dim1 hi | tile_dim0
  g1[4] = (int)(tile_d1 & 0xffffu);                      // tile_dim1 (2D)
  g1[5] = (int)stride_d0;                                // dim0_stride[31:0]
  g1[6] = 0;
  g1[7] = 0;
  const i32x4 z4 = {0, 0, 0, 0};
#if __clang_major__ >= 23
  const i32x8 z8 = {0, 0, 0, 0, 0, 0, 0, 0};
  __builtin_amdgcn_tensor_load_to_lds(g0, g1, z4, z4, z8, 0);
#else
  __builtin_amdgcn_tensor_load_to_lds(g0, g1, z4, z4, 0);
#endif
}
#endif  // HAS_TDM

// ---------------------------------------------------------------------------
// GEMM: C[M,N] = A[M,K] * W[K,N]
//   A_BF16:  A is bf16 in global (else f32, converted to bf16 in LDS)
//   OUT_F32: C written f32 (else bf16)
// Block: 256 threads (8 waves). Tile 128x128, BK=32.
// Wave grid 2(M) x 4(N): each wave computes 64x32 via 4x2 WMMA 16x16 tiles.
// ---------------------------------------------------------------------------
template <bool A_BF16, bool OUT_F32>
__global__ __launch_bounds__(256) void gemm_kernel(
    const void* __restrict__ Ap, const float* __restrict__ Ww,
    void* __restrict__ Cp, int M, int N, int K) {
  __shared__ __align__(16) __bf16 As[128][32];   // [m][k]
  __shared__ __align__(16) __bf16 Bs[128][32];   // [n][k]  (W transposed)

  const int tid  = threadIdx.x;
  const int lane = tid & 31;
  const int wid  = tid >> 5;
  const int lrow = lane & 15;
  const int hi   = lane >> 4;          // 0 / 1 half of wave
  const int wm   = (wid & 1) * 64;     // wave tile row offset
  const int wn   = (wid >> 1) * 32;    // wave tile col offset
  const int bm   = blockIdx.y * 128;
  const int bn   = blockIdx.x * 128;

  const v8f vzero = {0.f, 0.f, 0.f, 0.f, 0.f, 0.f, 0.f, 0.f};
  v8f acc[4][2];
  for (int i = 0; i < 4; ++i)
    for (int j = 0; j < 2; ++j) acc[i][j] = vzero;

  for (int kk = 0; kk < K; kk += 32) {
    // ---- stage A tile (128 x 32) ----
    if constexpr (A_BF16) {
      const __bf16* Ag = (const __bf16*)Ap;
#if HAS_ASYNC_B128
      for (int i = tid; i < 512; i += 256) {        // 8 halves per async op
        const int e = i * 8, r = e >> 5, c = e & 31;
        __builtin_amdgcn_global_load_async_to_lds_b128(
            (GlobV4*)(size_t)&Ag[(size_t)(bm + r) * K + kk + c],
            (LdsV4*)(unsigned)(size_t)&As[r][c],
            0, 0);
      }
#if HAS_WAIT_ASYNC
      __builtin_amdgcn_s_wait_asynccnt(0);
#else
      asm volatile("s_wait_asynccnt 0x0" ::: "memory");
#endif
#else
      for (int i = tid; i < 512; i += 256) {        // 8 halves per iter
        const int e = i * 8, r = e >> 5, c = e & 31;
        *(uint4*)&As[r][c] =
            *(const uint4*)&Ag[(size_t)(bm + r) * K + kk + c];
      }
#endif
      if (kk + 32 < K)
        __builtin_prefetch(&Ag[(size_t)(bm + (tid >> 1)) * K + kk + 32], 0, 1);
    } else {
      const float* Ag = (const float*)Ap;
      for (int i = tid; i < 1024; i += 256) {       // 4 floats per iter
        const int e = i * 4, r = e >> 5, c = e & 31;
        const float4 v = *(const float4*)&Ag[(size_t)(bm + r) * K + kk + c];
        As[r][c + 0] = (__bf16)v.x;  As[r][c + 1] = (__bf16)v.y;
        As[r][c + 2] = (__bf16)v.z;  As[r][c + 3] = (__bf16)v.w;
      }
      if (kk + 32 < K)
        __builtin_prefetch(&Ag[(size_t)(bm + (tid >> 1)) * K + kk + 32], 0, 1);
    }
    // ---- stage W tile (32 x 128), transposed into Bs[n][k] ----
    for (int i = tid; i < 1024; i += 256) {
      const int e = i * 4, kr = e >> 7, c = e & 127;
      const float4 v = *(const float4*)&Ww[(size_t)(kk + kr) * N + bn + c];
      Bs[c + 0][kr] = (__bf16)v.x;  Bs[c + 1][kr] = (__bf16)v.y;
      Bs[c + 2][kr] = (__bf16)v.z;  Bs[c + 3][kr] = (__bf16)v.w;
    }
    if (kk + 32 < K)
      __builtin_prefetch(&Ww[(size_t)(kk + 32 + (tid >> 3)) * N + bn], 0, 1);
    __syncthreads();

    // ---- fragments ----
    // A 16x32: lane<16 row=lrow elems K = hi*8+[0..7], 16+hi*8+[0..7]
    v16bf afr[4];
    for (int mf = 0; mf < 4; ++mf) {
      const __bf16* p = &As[wm + mf * 16 + lrow][hi * 8];
      afr[mf] = load_frag(p, p + 16);
    }
    // B 32x16: lane holds n=lrow, K = hi*8+[0..7], 16+hi*8+[0..7]
    v16bf bfr[2];
    for (int nf = 0; nf < 2; ++nf) {
      const __bf16* p = &Bs[wn + nf * 16 + lrow][hi * 8];
      bfr[nf] = load_frag(p, p + 16);
    }
    for (int mf = 0; mf < 4; ++mf)
      for (int nf = 0; nf < 2; ++nf)
        acc[mf][nf] = wmma_bf16(afr[mf], bfr[nf], acc[mf][nf]);
    __syncthreads();
  }

  // ---- epilogue: C layout: VGPR r -> row r (lanes 0-15) / r+8 (16-31) ----
  for (int mf = 0; mf < 4; ++mf)
    for (int nf = 0; nf < 2; ++nf)
      for (int r = 0; r < 8; ++r) {
        const int grow = bm + wm + mf * 16 + r + hi * 8;
        const int gcol = bn + wn + nf * 16 + lrow;
        const float val = acc[mf][nf][r];
        if constexpr (OUT_F32)
          ((float*)Cp)[(size_t)grow * N + gcol] = val;
        else
          ((__bf16*)Cp)[(size_t)grow * N + gcol] = (__bf16)val;
      }
}

// ---------------------------------------------------------------------------
// Windowed flash attention.
// Grid: B * H * (T/WIN) * (WIN/128) blocks of 256 threads (8 waves).
// Each wave: 16 queries x HD=128. Key loop: 32 keys/step over the
// [qb*WIN-WIN, qb*WIN+WIN) window with causal-window mask
//   qpos-WIN < kpos <= qpos.
// ---------------------------------------------------------------------------
__global__ __launch_bounds__(256) void attn_kernel(
    const __bf16* __restrict__ Q, const __bf16* __restrict__ Kb,
    const __bf16* __restrict__ Vb, __bf16* __restrict__ O) {
  __shared__ __align__(16) __bf16 Ks[32][128];     // [key][hd]
  __shared__ __align__(16) __bf16 Vt[128][32];     // [hd][key] (transposed)
  __shared__ __align__(16) __bf16 Ps[8][16][32];   // per-wave P tile [q][key]

  int idx = blockIdx.x;
  const int qchunk = idx & 3;  idx >>= 2;   // 128-query chunk in window block
  const int qb     = idx & 7;  idx >>= 3;   // window block (T/WIN = 8)
  const int h      = idx & 15; idx >>= 4;   // query head
  const int b      = idx;                   // batch
  const int hk     = h >> 2;                // kv head (nrep = 4)

  const int tid  = threadIdx.x;
  const int lane = tid & 31;
  const int wid  = tid >> 5;
  const int lrow = lane & 15;
  const int hi   = lane >> 4;

  const int    qbase = qb * WIN + qchunk * 128 + wid * 16;  // t of wave row 0
  const size_t qm0   = (size_t)b * Tt + qbase;

  // Q fragments (A layout), hoisted: 4 chunks of K=32 covering HD=128
  v16bf qf[4];
  {
    const __bf16* qrow = &Q[(qm0 + lrow) * (size_t)(Hh * HD) + h * HD];
    for (int kf = 0; kf < 4; ++kf) {
      const __bf16* p = qrow + kf * 32 + hi * 8;
      qf[kf] = load_frag(p, p + 16);
    }
  }

  const v8f vzero = {0.f, 0.f, 0.f, 0.f, 0.f, 0.f, 0.f, 0.f};
  v8f oacc[8];
  for (int nt = 0; nt < 8; ++nt) oacc[nt] = vzero;
  float mrun[8], lrun[8];
  for (int r = 0; r < 8; ++r) { mrun[r] = -1e30f; lrun[r] = 0.f; }

  const int   kstart = qb * WIN - WIN;
  const int   s0     = (qb == 0) ? 16 : 0;          // skip kpos < 0
  const int   s1     = 20 + 4 * qchunk;             // last step with kpos<=qmax
  const float scale  = 0.08838834764831845f;        // 1/sqrt(128)

  for (int s = s0; s < s1; ++s) {
    const int kpos0 = kstart + s * 32;
    const size_t gk = ((size_t)b * Tt + kpos0) * (HKk * HD) + hk * HD;

    // ---- stage K tile: TDM when available, else manual b128 copies ----
#if HAS_TDM
    if (wid == 0) {
      tdm_load_2d_bf16((unsigned)(size_t)&Ks[0][0], &Kb[gk],
                       /*tile_d0=*/HD, /*tile_d1=*/32,
                       /*tensor_d0=*/HKk * HD, /*tensor_d1=*/Bb * Tt,
                       /*stride_d0=*/HKk * HD);
      __builtin_amdgcn_s_wait_tensorcnt(0);
    }
#endif
    // ---- stage V tile transposed (manual: needs layout change) ----
    {
      const int    krow = tid >> 3, c = (tid & 7) * 16;
      const size_t g    = gk + (size_t)krow * (HKk * HD) + c;
#if !HAS_TDM
      const uint4* ks = (const uint4*)&Kb[g];
      *(uint4*)&Ks[krow][c]     = ks[0];
      *(uint4*)&Ks[krow][c + 8] = ks[1];
#endif
      BFrag vv;
      vv.u[0] = ((const uint4*)&Vb[g])[0];
      vv.u[1] = ((const uint4*)&Vb[g])[1];
      for (int j = 0; j < 16; ++j) Vt[c + j][krow] = vv.v[j];
    }
    __syncthreads();

    // ---- scores S = Q * K^T  (two 16x16 tiles over 32 keys) ----
    v8f sac[2] = {vzero, vzero};
    for (int t2 = 0; t2 < 2; ++t2)
      for (int kf = 0; kf < 4; ++kf) {
        const __bf16* p = &Ks[t2 * 16 + lrow][kf * 32 + hi * 8];
        sac[t2] = wmma_bf16(qf[kf], load_frag(p, p + 16), sac[t2]);
      }

    // ---- online softmax update ----
    float    sval[2][8];
    unsigned okm[2] = {0u, 0u};
    float    mt[8];
    for (int r = 0; r < 8; ++r) mt[r] = -1e30f;
    for (int t2 = 0; t2 < 2; ++t2) {
      const int kp = kpos0 + t2 * 16 + lrow;  // this lane's key column
      for (int r = 0; r < 8; ++r) {
        const int  qp = qbase + r + hi * 8;
        float      sv = sac[t2][r] * scale;
        const bool ok = (kp <= qp) && (kp + WIN > qp);
        sv            = ok ? sv : -1e30f;
        if (ok) okm[t2] |= (1u << r);
        sval[t2][r] = sv;
        mt[r]       = fmaxf(mt[r], sv);
      }
    }
    float alpha[8], rs[8];
    for (int r = 0; r < 8; ++r) {
      float m = mt[r];
      m = fmaxf(m, __shfl_xor(m, 1, 32));
      m = fmaxf(m, __shfl_xor(m, 2, 32));
      m = fmaxf(m, __shfl_xor(m, 4, 32));
      m = fmaxf(m, __shfl_xor(m, 8, 32));
      const float mnew = fmaxf(mrun[r], m);
      alpha[r]         = __expf(mrun[r] - mnew);
      mrun[r]          = mnew;
      rs[r]            = 0.f;
    }
    for (int t2 = 0; t2 < 2; ++t2)
      for (int r = 0; r < 8; ++r) {
        const float p =
            ((okm[t2] >> r) & 1u) ? __expf(sval[t2][r] - mrun[r]) : 0.f;
        rs[r] += p;
        Ps[wid][r + hi * 8][t2 * 16 + lrow] = (__bf16)p;
      }
    for (int r = 0; r < 8; ++r) {
      float t = rs[r];
      t += __shfl_xor(t, 1, 32);
      t += __shfl_xor(t, 2, 32);
      t += __shfl_xor(t, 4, 32);
      t += __shfl_xor(t, 8, 32);
      lrun[r] = lrun[r] * alpha[r] + t;
    }
    for (int nt = 0; nt < 8; ++nt)
      for (int r = 0; r < 8; ++r) oacc[nt][r] *= alpha[r];

    __syncthreads();  // Ps stores visible before A-fragment reads

    // ---- O += P * V  (A = P 16x32, B = V 32x16 per hd tile) ----
    const __bf16* pp = &Ps[wid][lrow][hi * 8];
    const v16bf   pa = load_frag(pp, pp + 16);
    for (int nt = 0; nt < 8; ++nt) {
      const __bf16* vp = &Vt[nt * 16 + lrow][hi * 8];
      oacc[nt] = wmma_bf16(pa, load_frag(vp, vp + 16), oacc[nt]);
    }
    __syncthreads();  // protect Ks/Vt before next staging
  }

  // ---- finalize: O /= l, write bf16 ----
  for (int r = 0; r < 8; ++r) {
    const float  inv  = 1.f / lrun[r];
    const int    trow = qbase + r + hi * 8;
    const size_t orow = ((size_t)b * Tt + trow) * (Hh * HD) + h * HD;
    for (int nt = 0; nt < 8; ++nt)
      O[orow + nt * 16 + lrow] = (__bf16)(oacc[nt][r] * inv);
  }
}

// ---------------------------------------------------------------------------
extern "C" void kernel_launch(void* const* d_in, const int* in_sizes, int n_in,
                              void* d_out, int out_size, void* d_ws,
                              size_t ws_size, hipStream_t stream) {
  (void)in_sizes; (void)n_in; (void)out_size; (void)ws_size;
  const float* x  = (const float*)d_in[0];
  const float* Wq = (const float*)d_in[1];
  const float* Wk = (const float*)d_in[2];
  const float* Wv = (const float*)d_in[3];
  const float* Wo = (const float*)d_in[4];

  char* ws = (char*)d_ws;
  const size_t qbytes  = (size_t)MTOT * (Hh * HD) * sizeof(__bf16);   // 32 MB
  const size_t kvbytes = (size_t)MTOT * (HKk * HD) * sizeof(__bf16);  //  8 MB
  __bf16* Qbf = (__bf16*)(ws);
  __bf16* Kbf = (__bf16*)(ws + qbytes);
  __bf16* Vbf = (__bf16*)(ws + qbytes + kvbytes);
  __bf16* Obf = (__bf16*)(ws + qbytes + 2 * kvbytes);

  const dim3 blk(256);
  // Projections: x[8192,2048] @ W -> bf16
  gemm_kernel<false, false><<<dim3(16, 64), blk, 0, stream>>>(
      x, Wq, Qbf, MTOT, Hh * HD, Dd);
  gemm_kernel<false, false><<<dim3(4, 64), blk, 0, stream>>>(
      x, Wk, Kbf, MTOT, HKk * HD, Dd);
  gemm_kernel<false, false><<<dim3(4, 64), blk, 0, stream>>>(
      x, Wv, Vbf, MTOT, HKk * HD, Dd);
  // Attention: B*H*(T/WIN)*(WIN/128) = 2*16*8*4 = 1024 blocks
  attn_kernel<<<dim3(Bb * Hh * (Tt / WIN) * (WIN / 128)), blk, 0, stream>>>(
      Qbf, Kbf, Vbf, Obf);
  // Output projection: O[8192,2048] @ Wo -> f32 out
  gemm_kernel<true, true><<<dim3(16, 64), blk, 0, stream>>>(
      Obf, Wo, (float*)d_out, MTOT, Dd, Hh * HD);
}
